// StatefulRNN_51299089383928
// MI455X (gfx1250) — matrix operand order repaired and checked by reading, not compiled, not run on hardware
//
#include <hip/hip_runtime.h>

// ---------------------------------------------------------------------------
// GRU (reset_after) : x_proj GEMM (bf16 WMMA, 128x128 tiles) + persistent scan
// kernel with per-WGP-resident Wr slab in 320KB LDS and per-step grid barrier.
// T=512, B=256, D=256, U=512.
// ---------------------------------------------------------------------------

typedef __attribute__((ext_vector_type(16))) __bf16 v16bf;
typedef __attribute__((ext_vector_type(8)))  __bf16 v8bf;
typedef __attribute__((ext_vector_type(8)))  float  v8f;

#define T_STEPS 512
#define BATCH   256
#define DIN     256
#define UNITS   512
#define N3U     1536
#define APITCH  40    // 32 bf16 payload + 8 pad (80B rows -> conflict-free b128)
#define BPITCH  520   // 512 bf16 payload + 8 pad (1040B rows -> 4L mod 64 banks)

// ---- fragment loaders (ISA 7.12.2 layouts, wave32) -------------------------
// A (16x32 bf16): lane L<16 -> row M=L, K={8h..8h+7, 16+8h..+7}, h=lane>=16
__device__ __forceinline__ v16bf load_a_frag(const __bf16* As, int rowBase, int lane) {
    int row = rowBase + (lane & 15);
    int kh  = lane >> 4;
    const __bf16* p = As + row * APITCH;
    v8bf lo = *(const v8bf*)(p + 8 * kh);
    v8bf hi = *(const v8bf*)(p + 16 + 8 * kh);
    v16bf a;
#pragma unroll
    for (int i = 0; i < 8; ++i) { a[i] = lo[i]; a[8 + i] = hi[i]; }
    return a;
}

// B (32x16 bf16), stored N-major (base[n][k], given pitch): lane -> N=lane&15,
// K = k0 + {8h..8h+7, 16+8h..+7}, h=lane>=16.
__device__ __forceinline__ v16bf load_b_frag(const __bf16* base, int n0, int k0,
                                             int lane, int pitch) {
    int n  = n0 + (lane & 15);
    int kh = lane >> 4;
    const __bf16* p = base + n * pitch + k0 + 8 * kh;
    v8bf lo = *(const v8bf*)(p);
    v8bf hi = *(const v8bf*)(p + 16);
    v16bf b;
#pragma unroll
    for (int i = 0; i < 8; ++i) { b[i] = lo[i]; b[8 + i] = hi[i]; }
    return b;
}

// ---------------------------------------------------------------------------
// Kernel 0: weight transpose+convert to bf16, h init, barrier init.
// WrT[n][k] (n<1536,k<512), WxT[n][k] (n<1536,k<256)
// ---------------------------------------------------------------------------
__global__ __launch_bounds__(256) void prep_kernel(
        const float* __restrict__ Wx, const float* __restrict__ Wr,
        const float* __restrict__ h0, __bf16* __restrict__ WrT,
        __bf16* __restrict__ WxT, float* __restrict__ hbuf,
        unsigned* __restrict__ syncCnt, unsigned* __restrict__ syncGen) {
    const int NWR = UNITS * N3U;   // 786432
    const int NWX = DIN * N3U;     // 393216
    const int NH  = BATCH * UNITS; // 131072
    const int total = NWR + NWX + NH;
    int stride = gridDim.x * blockDim.x;
    for (int i = blockIdx.x * blockDim.x + threadIdx.x; i < total; i += stride) {
        if (i < NWR) {
            int n = i / UNITS, k = i - n * UNITS;
            WrT[i] = (__bf16)Wr[(size_t)k * N3U + n];
        } else if (i < NWR + NWX) {
            int j = i - NWR;
            int n = j / DIN, k = j - n * DIN;
            WxT[j] = (__bf16)Wx[(size_t)k * N3U + n];
        } else {
            int j = i - NWR - NWX;
            hbuf[j] = h0[j];
        }
    }
    if (blockIdx.x == 0 && threadIdx.x == 0) { *syncCnt = 0u; *syncGen = 0u; }
}

// ---------------------------------------------------------------------------
// Kernel 1: x_proj = X[T*B, 256] @ Wx[256, 1536] + bias0 -> bf16 XP.
// 128x128 block tile, 8 waves; wave = 32(M) x 64(N): 2 A-frags + 4 B-frags
// feed 8 WMMAs per 32-wide K-tile.
// ---------------------------------------------------------------------------
__global__ __launch_bounds__(256) void xproj_gemm(
        const float* __restrict__ X, const __bf16* __restrict__ WxT,
        const float* __restrict__ bias0, __bf16* __restrict__ XP) {
    __shared__ __bf16 As[128 * APITCH];
    __shared__ __bf16 Bs[128 * APITCH];

    const int tid  = threadIdx.x;
    const int lane = tid & 31, w = tid >> 5;
    const int mq = w & 3;          // 4 x 32-row M strips
    const int nh = w >> 2;         // 2 x 64-col N strips
    const long mBase = (long)blockIdx.x * 128;
    const int  nBase = blockIdx.y * 128;

    v8f acc[2][4];
#pragma unroll
    for (int mi = 0; mi < 2; ++mi)
#pragma unroll
        for (int s = 0; s < 4; ++s) acc[mi][s] = (v8f){};

    for (int k0 = 0; k0 < DIN; k0 += 32) {
        __syncthreads();
        // A tile: 128 rows x 32 cols fp32 -> bf16 (1024 float4 chunks)
#pragma unroll
        for (int q = 0; q < 4; ++q) {
            int id = tid + 256 * q;
            int row = id >> 3, c4 = (id & 7) * 4;
            float4 v = *(const float4*)(X + (mBase + row) * DIN + k0 + c4);
            __bf16* d = As + row * APITCH + c4;
            d[0] = (__bf16)v.x; d[1] = (__bf16)v.y;
            d[2] = (__bf16)v.z; d[3] = (__bf16)v.w;
        }
        // B tile: 128 n-rows x 32 k (512 v8bf chunks)
#pragma unroll
        for (int q = 0; q < 2; ++q) {
            int id = tid + 256 * q;
            int n = id >> 2, c8 = (id & 3) * 8;
            *(v8bf*)(Bs + n * APITCH + c8) =
                *(const v8bf*)(WxT + (size_t)(nBase + n) * DIN + k0 + c8);
        }
        __syncthreads();
        v16bf a0 = load_a_frag(As, mq * 32, lane);
        v16bf a1 = load_a_frag(As, mq * 32 + 16, lane);
#pragma unroll
        for (int s = 0; s < 4; ++s) {
            v16bf b = load_b_frag(Bs, nh * 64 + s * 16, 0, lane, APITCH);
            acc[0][s] = __builtin_amdgcn_wmma_f32_16x16x32_bf16(
                    false, a0, false, b, (short)0, acc[0][s], false, false);
            acc[1][s] = __builtin_amdgcn_wmma_f32_16x16x32_bf16(
                    false, a1, false, b, (short)0, acc[1][s], false, false);
        }
    }

    const int laneHi = lane >> 4, nl = lane & 15;
#pragma unroll
    for (int s = 0; s < 4; ++s) {
        int n = nBase + nh * 64 + s * 16 + nl;
        float bv = bias0[n];
#pragma unroll
        for (int mi = 0; mi < 2; ++mi)
#pragma unroll
            for (int j = 0; j < 8; ++j) {
                long m = mBase + mq * 32 + mi * 16 + j + 8 * laneHi;
                XP[m * N3U + n] = (__bf16)(acc[mi][s][j] + bv);
            }
    }
}

// ---------------------------------------------------------------------------
// Kernel 2: persistent GRU scan.  32 blocks (4 M-tiles x 8 U-tiles of 64).
// Wr slab (3 gates x 64 N x 512 K bf16 ~ 195KB) loaded ONCE into LDS and kept
// resident for all 512 steps (CDNA5 320KB LDS).  Masked-h A tiles (5KB) are
// double-buffered -> 1 barrier per K-tile.  Grid barrier per timestep.
// ---------------------------------------------------------------------------
__global__ __launch_bounds__(256) void gru_scan(
        const __bf16* __restrict__ XP, const __bf16* __restrict__ WrT,
        const float* __restrict__ bias1, const unsigned char* __restrict__ mask,
        float* __restrict__ h0buf, float* __restrict__ h1buf,
        float* __restrict__ out,
        unsigned* __restrict__ syncCnt, unsigned* __restrict__ syncGen) {
    __shared__ __bf16 BsAll[3 * 64 * BPITCH];   // 199,680 B, persistent
    __shared__ __bf16 As[2][64 * APITCH];       //  10,240 B, double-buffered

    const int tid  = threadIdx.x;
    const int lane = tid & 31, w = tid >> 5;
    const int mw2 = w & 1;         // 2 x 32-row M strips
    const int nq  = w >> 1;        // 4 x 16-col U strips
    const int mBase = (blockIdx.x & 3) * 64;
    const int uBase = (blockIdx.x >> 2) * 64;
    const unsigned nBlocks = gridDim.x;
    unsigned localGen = 0;

    // ---- one-time resident Wr slab load: 3 x 64 x 512 bf16 -----------------
    for (int c = tid; c < 3 * 64 * 64; c += 256) {      // chunks of 8 bf16
        int g   = c / (64 * 64);
        int rem = c - g * (64 * 64);
        int n   = rem >> 6;
        int kc  = (rem & 63) * 8;
        *(v8bf*)(BsAll + (g * 64 + n) * BPITCH + kc) =
            *(const v8bf*)(WrT + (size_t)(g * UNITS + uBase + n) * UNITS + kc);
    }

    for (int t = 0; t < T_STEPS; ++t) {
        const float* hsrc = (t & 1) ? h1buf : h0buf;
        float*       hdst = (t & 1) ? h0buf : h1buf;
        const unsigned char* mrow = mask + (size_t)t * BATCH;

        auto loadA = [&](int k0, __bf16* dst) {
            int r = tid >> 3, c4 = (tid & 7) * 4;
#pragma unroll
            for (int rr = 0; rr < 2; ++rr) {
                int row = r + rr * 32;
                int m = mBase + row;
                float4 v = *(const float4*)(hsrc + (size_t)m * UNITS + k0 + c4);
                float msk = mrow[m] ? 1.f : 0.f;
                __bf16* d = dst + row * APITCH + c4;
                d[0] = (__bf16)(v.x * msk); d[1] = (__bf16)(v.y * msk);
                d[2] = (__bf16)(v.z * msk); d[3] = (__bf16)(v.w * msk);
            }
        };

        v8f acc[3][2];
#pragma unroll
        for (int g = 0; g < 3; ++g)
#pragma unroll
            for (int mi = 0; mi < 2; ++mi) acc[g][mi] = (v8f){};

        loadA(0, As[0]);                    // prime buffer 0
        const int NITER = UNITS / 32;       // 16
        for (int ki = 0; ki < NITER; ++ki) {
            __syncthreads();                // As[ki&1] ready; Bs slab + prev reads done
            if (ki + 1 < NITER)
                loadA((ki + 1) * 32, As[(ki + 1) & 1]);
            const __bf16* Ab = As[ki & 1];
            v16bf a0 = load_a_frag(Ab, mw2 * 32, lane);
            v16bf a1 = load_a_frag(Ab, mw2 * 32 + 16, lane);
#pragma unroll
            for (int g = 0; g < 3; ++g) {
                v16bf b = load_b_frag(BsAll + g * 64 * BPITCH, nq * 16,
                                      ki * 32, lane, BPITCH);
                acc[g][0] = __builtin_amdgcn_wmma_f32_16x16x32_bf16(
                        false, a0, false, b, (short)0, acc[g][0], false, false);
                acc[g][1] = __builtin_amdgcn_wmma_f32_16x16x32_bf16(
                        false, a1, false, b, (short)0, acc[g][1], false, false);
            }
        }

        // gates + state update
        const int laneHi = lane >> 4, nl = lane & 15;
        {
            int u = uBase + nq * 16 + nl;
            float brz = bias1[u];
            float brr = bias1[UNITS + u];
            float brh = bias1[2 * UNITS + u];
#pragma unroll
            for (int mi = 0; mi < 2; ++mi)
#pragma unroll
                for (int j = 0; j < 8; ++j) {
                    int m = mBase + mw2 * 32 + mi * 16 + j + 8 * laneHi;
                    size_t xpo = ((size_t)t * BATCH + m) * N3U + u;
                    float xz = (float)XP[xpo];
                    float xr = (float)XP[xpo + UNITS];
                    float xh = (float)XP[xpo + 2 * UNITS];
                    float hm = mrow[m] ? hsrc[(size_t)m * UNITS + u] : 0.f;
                    float z  = 1.f / (1.f + __expf(-(xz + acc[0][mi][j] + brz)));
                    float r  = 1.f / (1.f + __expf(-(xr + acc[1][mi][j] + brr)));
                    float hh = tanhf(xh + r * (acc[2][mi][j] + brh));
                    float hn = z * hm + (1.f - z) * hh;
                    hdst[(size_t)m * UNITS + u] = hn;
                    out[((size_t)t * BATCH + m) * UNITS + u] = hn;
                }
        }

        // ---- grid-wide sense-reversal barrier (32 resident blocks) --------
        __threadfence();
        __syncthreads();
        if (tid == 0) {
            unsigned old = atomicAdd(syncCnt, 1u);
            if (old == nBlocks - 1) {
                atomicExch(syncCnt, 0u);
                __threadfence();
                atomicAdd(syncGen, 1u);
            } else {
                while (atomicAdd(syncGen, 0u) <= localGen) {
                    __builtin_amdgcn_s_sleep(8);
                }
            }
        }
        localGen++;
        __syncthreads();
    }
}

// ---------------------------------------------------------------------------
extern "C" void kernel_launch(void* const* d_in, const int* in_sizes, int n_in,
                              void* d_out, int out_size, void* d_ws, size_t ws_size,
                              hipStream_t stream) {
    (void)in_sizes; (void)n_in; (void)out_size; (void)ws_size;
    const float*         X    = (const float*)d_in[0];
    const unsigned char* mask = (const unsigned char*)d_in[1]; // bool8
    const float*         Wx   = (const float*)d_in[2];
    const float*         Wr   = (const float*)d_in[3];
    const float*         bias = (const float*)d_in[4];         // [2,1536]
    const float*         h0   = (const float*)d_in[5];
    float* out = (float*)d_out;

    char* ws = (char*)d_ws;
    size_t off = 0;
    __bf16* XP   = (__bf16*)(ws + off); off += (size_t)T_STEPS * BATCH * N3U * 2;
    __bf16* WrT  = (__bf16*)(ws + off); off += (size_t)UNITS * N3U * 2;
    __bf16* WxT  = (__bf16*)(ws + off); off += (size_t)DIN * N3U * 2;
    float*  h0b  = (float*)(ws + off);  off += (size_t)BATCH * UNITS * 4;
    float*  h1b  = (float*)(ws + off);  off += (size_t)BATCH * UNITS * 4;
    unsigned* syncCnt = (unsigned*)(ws + off);
    unsigned* syncGen = syncCnt + 1;

    prep_kernel<<<1024, 256, 0, stream>>>(Wx, Wr, h0, WrT, WxT, h0b,
                                          syncCnt, syncGen);
    dim3 g1((T_STEPS * BATCH) / 128, N3U / 128);
    xproj_gemm<<<g1, 256, 0, stream>>>(X, WxT, bias, XP);
    gru_scan<<<32, 256, 0, stream>>>(XP, WrT, bias + N3U, mask, h0b, h1b, out,
                                     syncCnt, syncGen);
}
